// DNFNetLocalization_8589934592444
// MI455X (gfx1250) — matrix operand order repaired and checked by reading, not compiled, not run on hardware
//
#include <hip/hip_runtime.h>

// DNFNet localization: dist^2 = ||(x - mu) * sigma||^2 expanded into a
// K=512 bf16 WMMA GEMM (A = [x^2, x], B = [sigma^2, -2*mu*sigma^2], bias
// c_f = sum mu^2 sigma^2), fused with the exp/sigmoid-gate/softmax epilogue.
// Target: gfx1250 (CDNA5, wave32, v_wmma_f32_16x16x32_bf16).
// Rev 3: 4-slot rotating B-fragment buffer with prefetch distance 3 over the
// flattened (k,t) tile stream, so each WMMA waits on loadcnt<=6 instead of
// draining to zero (keeps ~3 tile-loads in flight over L2 latency).

typedef __attribute__((ext_vector_type(16))) __bf16        v16bf;
typedef __attribute__((ext_vector_type(8)))  float         v8f;
typedef __attribute__((ext_vector_type(4)))  unsigned int  v4u;

#define NUM_FORMULAS 1024
#define INPUT_DIM    256
#define BATCH        2048
#define KDIM         512   // 256 (x^2 terms) + 256 (x terms)

__device__ __forceinline__ unsigned short f2bf(float f) {
  union { float f; unsigned int u; } x; x.f = f;
  unsigned int r = x.u + 0x7FFFu + ((x.u >> 16) & 1u);   // round-to-nearest-even
  return (unsigned short)(r >> 16);
}

// ---------------------------------------------------------------------------
// Prep 1: A_pk[b][0:256] = bf16(x^2), A_pk[b][256:512] = bf16(x). Row-major.
// ---------------------------------------------------------------------------
__global__ __launch_bounds__(256) void pack_A_kernel(
    const float* __restrict__ x, unsigned short* __restrict__ A_pk) {
  int i = blockIdx.x * 256 + threadIdx.x;       // over BATCH*KDIM
  int b = i >> 9;
  int j = i & (KDIM - 1);
  float v;
  if (j < INPUT_DIM) {
    float xv = x[b * INPUT_DIM + j];
    v = xv * xv;
  } else {
    v = x[b * INPUT_DIM + (j - INPUT_DIM)];
  }
  A_pk[i] = f2bf(v);
}

// ---------------------------------------------------------------------------
// Prep 2: per formula f: B_pk[f][0:256] = bf16(sigma^2),
//         B_pk[f][256:512] = bf16(-2*mu*sigma^2); c[f] = sum mu^2 sigma^2.
// One 256-thread block per formula (column-contiguous over K for WMMA B).
// ---------------------------------------------------------------------------
__global__ __launch_bounds__(256) void pack_B_kernel(
    const float* __restrict__ mu, const float* __restrict__ sigma,
    unsigned short* __restrict__ B_pk, float* __restrict__ cvec) {
  int f = blockIdx.x;
  int d = threadIdx.x;
  float s  = sigma[f * INPUT_DIM + d];
  float m  = mu[f * INPUT_DIM + d];
  float s2 = s * s;
  B_pk[f * KDIM + d]             = f2bf(s2);
  B_pk[f * KDIM + INPUT_DIM + d] = f2bf(-2.0f * m * s2);

  __shared__ float red[256];
  red[d] = m * m * s2;
  __syncthreads();
  for (int st = 128; st > 0; st >>= 1) {
    if (d < st) red[d] += red[d + st];
    __syncthreads();
  }
  if (d == 0) cvec[f] = red[0];
}

// ---------------------------------------------------------------------------
// Main: 16 batch rows per workgroup, 8 waves x 128 formulas each.
// K-loop of v_wmma_f32_16x16x32_bf16, fused sqrt/exp/softmax epilogue.
// ---------------------------------------------------------------------------
__global__ __launch_bounds__(256) void dnf_wmma_softmax_kernel(
    const unsigned short* __restrict__ A_pk,
    const unsigned short* __restrict__ B_pk,
    const float* __restrict__ cvec,
    const float* __restrict__ temperature,
    float* __restrict__ out) {
  const int m0   = blockIdx.x * 16;       // batch-row tile
  const int tid  = threadIdx.x;
  const int wid  = tid >> 5;              // wave id 0..7
  const int lane = tid & 31;
  const int half = lane >> 4;             // 0: lanes 0-15, 1: lanes 16-31
  const int l16  = lane & 15;
  const int n0   = wid * 128;             // this wave's formula base

  // --- accumulators: 8 N-tiles of 16x16 f32, init to bias c_f per column ---
  v8f acc[8];
  #pragma unroll
  for (int t = 0; t < 8; ++t) {
    float cv = cvec[n0 + t * 16 + l16];   // C/D layout: N = lane%16
    #pragma unroll
    for (int v = 0; v < 8; ++v) acc[t][v] = cv;
  }

  union Frag { v16bf v; v4u q[2]; };

  // A fragment (bf16 16x32): lane<16 holds row M=l16, K 0..15;
  // lane>=16 same row, K 16..31.
  const unsigned short* arow = A_pk + (size_t)(m0 + l16) * KDIM;
  const int a_off0 = half * 16;           // elements 0..7
  const int a_off1 = half * 16 + 8;       // elements 8..15
  // B fragment (bf16 32x16): lane<16 holds col N=l16, K 0..15; lane>=16 K 16..31.
  const unsigned short* bbase = B_pk + (size_t)(n0 + l16) * KDIM + half * 16;
  const size_t bcolstride = (size_t)16 * KDIM;   // advance 16 formulas

  Frag fa[2];
  Frag fb[4];                             // 4-slot rotating buffer, PF dist 3

  // tile index j in [0,128): k-step = j>>3, n-tile t = j&7
  // ---- prologue: A(k=0) and B tiles j=0,1,2 ----
  fa[0].q[0] = *(const v4u*)(arow + a_off0);
  fa[0].q[1] = *(const v4u*)(arow + a_off1);
  #pragma unroll
  for (int j = 0; j < 3; ++j) {
    const unsigned short* bp = bbase + (size_t)j * bcolstride;  // k=0, t=j
    fb[j].q[0] = *(const v4u*)(bp);
    fb[j].q[1] = *(const v4u*)(bp + 8);
  }

  #pragma unroll
  for (int ks = 0; ks < KDIM / 32; ++ks) {
    const int k0 = ks * 32;
    const int ka = ks & 1;
    // issue next k-step's A fragment loads (consumed 8 tiles later)
    if (k0 + 32 < KDIM) {
      fa[ka ^ 1].q[0] = *(const v4u*)(arow + (k0 + 32) + a_off0);
      fa[ka ^ 1].q[1] = *(const v4u*)(arow + (k0 + 32) + a_off1);
      __builtin_prefetch(arow + k0 + 64, 0, 3);
    }
    #pragma unroll
    for (int t = 0; t < 8; ++t) {
      const int j  = ks * 8 + t;
      const int jn = j + 3;               // prefetch distance 3 tiles
      if (jn < 128) {
        const int kn = (jn >> 3) * 32;
        const int tn = jn & 7;
        const unsigned short* bp = bbase + (size_t)tn * bcolstride + kn;
        fb[jn & 3].q[0] = *(const v4u*)(bp);
        fb[jn & 3].q[1] = *(const v4u*)(bp + 8);
      }
      acc[t] = __builtin_amdgcn_wmma_f32_16x16x32_bf16(
          /*neg_a=*/false, fa[ka].v, /*neg_b=*/false, fb[j & 3].v,
          /*c_mod=*/(short)0, acc[t], /*reuse_a=*/false, /*reuse_b=*/false);
    }
  }

  // --- epilogue: dist = sqrt(max(d2,0)); e = exp(sig(T)*exp(-dist)) ---
  const float T = temperature[0];
  const float gate = 1.0f / (1.0f + __expf(-T));  // sigmoid(T)

  float partial[8];
  #pragma unroll
  for (int v = 0; v < 8; ++v) partial[v] = 0.0f;

  #pragma unroll
  for (int t = 0; t < 8; ++t) {
    #pragma unroll
    for (int v = 0; v < 8; ++v) {
      float d2  = acc[t][v];
      float d   = __builtin_sqrtf(fmaxf(d2, 0.0f));
      float loc = __expf(-d);
      float e   = __expf(gate * loc);   // gated values < 1 => no max-shift needed
      acc[t][v] = e;                    // reuse accumulator regs for exp values
      partial[v] += e;
    }
  }

  // reduce across the 16 lanes that share the same row (width-16 xor shuffle)
  #pragma unroll
  for (int m = 1; m < 16; m <<= 1) {
    #pragma unroll
    for (int v = 0; v < 8; ++v)
      partial[v] += __shfl_xor(partial[v], m, 16);
  }

  // cross-wave reduction: each wave contributes its 128-column partial sums
  __shared__ float wsum[8][16];
  __shared__ float rowtotal[16];
  if (l16 == 0) {
    #pragma unroll
    for (int v = 0; v < 8; ++v) wsum[wid][half * 8 + v] = partial[v];
  }
  __syncthreads();
  if (tid < 16) {
    float s = 0.0f;
    #pragma unroll
    for (int w = 0; w < 8; ++w) s += wsum[w][tid];
    rowtotal[tid] = s;
  }
  __syncthreads();

  // normalize and store: out[(m0+row), n0 + 16t + l16]
  #pragma unroll
  for (int v = 0; v < 8; ++v) {
    int row_local = v + 8 * half;
    float inv = 1.0f / rowtotal[row_local];
    float* orow = out + (size_t)(m0 + row_local) * NUM_FORMULAS;
    #pragma unroll
    for (int t = 0; t < 8; ++t) {
      orow[n0 + t * 16 + l16] = acc[t][v] * inv;
    }
  }
}

// ---------------------------------------------------------------------------
extern "C" void kernel_launch(void* const* d_in, const int* in_sizes, int n_in,
                              void* d_out, int out_size, void* d_ws, size_t ws_size,
                              hipStream_t stream) {
  const float* x     = (const float*)d_in[0];   // (2048, 256)
  const float* mu    = (const float*)d_in[1];   // (1024, 256)
  const float* sigma = (const float*)d_in[2];   // (1, 1024, 256)
  const float* temp  = (const float*)d_in[3];   // scalar
  float* out = (float*)d_out;                   // (2048, 1024)

  unsigned short* A_pk = (unsigned short*)d_ws;                       // 2 MB
  unsigned short* B_pk = A_pk + (size_t)BATCH * KDIM;                 // 1 MB
  float* cvec = (float*)(B_pk + (size_t)NUM_FORMULAS * KDIM);         // 4 KB

  pack_A_kernel<<<(BATCH * KDIM) / 256, 256, 0, stream>>>(x, A_pk);
  pack_B_kernel<<<NUM_FORMULAS, 256, 0, stream>>>(mu, sigma, B_pk, cvec);
  dnf_wmma_softmax_kernel<<<BATCH / 16, 256, 0, stream>>>(A_pk, B_pk, cvec,
                                                          temp, out);
}
